// GeometricMatchingCore_12764642803851
// MI455X (gfx1250) — matrix operand order repaired and checked by reading, not compiled, not run on hardware
//
#include <hip/hip_runtime.h>
#include <hip/hip_bf16.h>

typedef __attribute__((ext_vector_type(16))) _Float16 v16h;
typedef __attribute__((ext_vector_type(8)))  float    v8f;

#define H_IMG 48
#define W_IMG 48
#define NPIX  2304            // 48*48
#define SEQ   2304
#define EMB   512
#define FF_DIM 2048
#define NHEAD 8
#define HDIM  64

#if __has_builtin(__builtin_amdgcn_global_load_async_to_lds_b128)
#define HAS_ASYNC_LDS 1
#warning "CDNA5: using global_load_async_to_lds_b128 path"
#else
#define HAS_ASYNC_LDS 0
#warning "CDNA5: async-to-LDS builtin unavailable, using sync fallback"
#endif

#if HAS_ASYNC_LDS
typedef int v4i_vec __attribute__((vector_size(16)));
typedef __attribute__((address_space(1))) v4i_vec glb_v4i;
typedef __attribute__((address_space(3))) v4i_vec lds_v4i;
static __device__ __forceinline__ void async_cp16(const _Float16* g, _Float16* l) {
  __builtin_amdgcn_global_load_async_to_lds_b128((glb_v4i*)g, (lds_v4i*)l, 0, 0);
}
static __device__ __forceinline__ void wait_async0() {
#if __has_builtin(__builtin_amdgcn_s_wait_asynccnt)
  __builtin_amdgcn_s_wait_asynccnt(0);
#else
  asm volatile("s_wait_asynccnt 0x0" ::: "memory");
#endif
}
#endif

// ---------------------------------------------------------------------------
// small utility kernels
// ---------------------------------------------------------------------------
__global__ void concat_kernel(const float* __restrict__ a, const float* __restrict__ b,
                              float* __restrict__ o) {
  int i = blockIdx.x * blockDim.x + threadIdx.x;
  if (i < 3 * NPIX) { o[i] = a[i]; o[3 * NPIX + i] = b[i]; }
}

// fp32 (K,N) row-major  ->  f16 (N,K) row-major (transposed weight for GEMM B)
__global__ void f2h_t_kernel(const float* __restrict__ a, _Float16* __restrict__ o,
                             int K, int N) {
  int idx = blockIdx.x * blockDim.x + threadIdx.x;
  if (idx >= K * N) return;
  int n = idx / K, k = idx - n * K;
  o[idx] = (_Float16)a[(size_t)k * N + n];
}

// direct 3x3 conv, pad=1, NCHW, OIHW weights. act: 0 none, 1 relu, 2 tanh, 3 sigmoid
__global__ void conv3x3_kernel(const float* __restrict__ in, const float* __restrict__ w,
                               const float* __restrict__ bias, float* __restrict__ out,
                               int Cin, int Cout, int act) {
  int idx = blockIdx.x * blockDim.x + threadIdx.x;
  if (idx >= Cout * NPIX) return;
  int co = idx / NPIX;
  int p  = idx - co * NPIX;
  int y  = p / W_IMG, x = p - (p / W_IMG) * W_IMG;
  float acc = bias[co];
  const float* wp = w + (size_t)co * Cin * 9;
  for (int ci = 0; ci < Cin; ++ci) {
    const float* ip = in + (size_t)ci * NPIX;
    const float* wc = wp + ci * 9;
#pragma unroll
    for (int ky = 0; ky < 3; ++ky) {
      int yy = y + ky - 1;
      if (yy < 0 || yy > 47) continue;
#pragma unroll
      for (int kx = 0; kx < 3; ++kx) {
        int xx = x + kx - 1;
        if (xx < 0 || xx > 47) continue;
        acc = __builtin_fmaf(ip[yy * W_IMG + xx], wc[ky * 3 + kx], acc);
      }
    }
  }
  if (act == 1) acc = fmaxf(acc, 0.0f);
  else if (act == 2) acc = tanhf(acc);
  else if (act == 3) acc = 1.0f / (1.0f + __expf(-acc));
  out[idx] = acc;
}

// t[s][d] = c3[d][s] + posenc(d, h(s));  also emits f16 copy
__global__ void posenc_transpose_kernel(const float* __restrict__ c3,
                                        float* __restrict__ t, _Float16* __restrict__ th) {
  int idx = blockIdx.x * blockDim.x + threadIdx.x;
  if (idx >= SEQ * EMB) return;
  int s = idx / EMB, d = idx - s * EMB;
  int h = s / W_IMG;
  float expo = (float)(d & ~1) / (float)EMB;
  float ang  = (float)h / __powf(10000.0f, expo);
  float pe   = (d & 1) ? __cosf(ang) : __sinf(ang);
  float v = c3[(size_t)d * NPIX + s] + pe;
  t[idx]  = v;
  th[idx] = (_Float16)v;
}

__global__ void feat_transpose_kernel(const float* __restrict__ t, float* __restrict__ f) {
  int idx = blockIdx.x * blockDim.x + threadIdx.x;
  if (idx >= SEQ * EMB) return;
  int d = idx / NPIX, s = idx - d * NPIX;
  f[idx] = t[(size_t)s * EMB + d];
}

// ---------------------------------------------------------------------------
// WMMA GEMM: C = act(A(MxK,f16) * BT(NxK,f16)^T + bias), fp32 accumulate.
// BM=64 BN=128 BK=32; 256 threads = 8 waves (2x4); each wave owns a 32x32
// output tile = 4 WMMA per k-step. A/B tiles double-buffered in LDS and
// filled with GLOBAL_LOAD_ASYNC_TO_LDS_B128 (ASYNCcnt) when available.
// ---------------------------------------------------------------------------
#define GBM 64
#define GBN 128
#define GBK 32
#define AST 48     // padded LDS stride (halves): 96B rows, 16B-aligned
#define BST 48

__global__ __launch_bounds__(256)
void gemm_wmma_kernel(const _Float16* __restrict__ A, const _Float16* __restrict__ BT,
                      const float* __restrict__ bias, float* __restrict__ C32,
                      _Float16* __restrict__ C16, int M, int N, int K, int act) {
  __shared__ __align__(16) _Float16 As[2][GBM * AST];
  __shared__ __align__(16) _Float16 Bs[2][GBN * BST];
  const int tid  = threadIdx.x;
  const int lane = tid & 31;
  const int wid  = tid >> 5;
  const int m0 = blockIdx.y * GBM;
  const int n0 = blockIdx.x * GBN;
  const int tm = (wid & 1) * 32;
  const int tn = (wid >> 1) * 32;
  const int lrow = lane & 15;
  const int hid  = lane >> 4;

  // cooperative tile-load coordinates
  const int arow = tid >> 2;          // 0..63
  const int akg  = (tid & 3) * 8;     // 8-half (16B) chunks of the 32-wide k tile
  const int brow = tid >> 1;          // 0..127
  const int bkg  = (tid & 1) * 16;    // 16-half chunk -> two 16B pieces

  const _Float16* aSrc = A  + (size_t)(m0 + arow) * K + akg;
  const _Float16* bSrc = BT + (size_t)(n0 + brow) * K + bkg;

  auto stage = [&](int buf, int k0) {
#if HAS_ASYNC_LDS
    async_cp16(aSrc + k0, &As[buf][arow * AST + akg]);
    async_cp16(bSrc + k0,     &Bs[buf][brow * BST + bkg]);
    async_cp16(bSrc + k0 + 8, &Bs[buf][brow * BST + bkg + 8]);
#else
    *(uint4*)&As[buf][arow * AST + akg]     = *(const uint4*)(aSrc + k0);
    *(uint4*)&Bs[buf][brow * BST + bkg]     = *(const uint4*)(bSrc + k0);
    *(uint4*)&Bs[buf][brow * BST + bkg + 8] = *(const uint4*)(bSrc + k0 + 8);
#endif
  };

  v8f acc00 = {}, acc01 = {}, acc10 = {}, acc11 = {};
  stage(0, 0);
#if HAS_ASYNC_LDS
  wait_async0();
#endif
  __syncthreads();

  const int T = K / GBK;
  for (int i = 0; i < T; ++i) {
    const int cur = i & 1;
    if (i + 1 < T) stage(1 - cur, (i + 1) * GBK);   // DMA next tile during math

    v16h a0, a1, b0, b1;
    const _Float16* ar0 = &As[cur][(tm + lrow) * AST];
    const _Float16* ar1 = &As[cur][(tm + 16 + lrow) * AST];
    const _Float16* br0 = &Bs[cur][(tn + lrow) * BST];
    const _Float16* br1 = &Bs[cur][(tn + 16 + lrow) * BST];
#pragma unroll
    for (int vg = 0; vg < 8; ++vg) {
      int ka = hid * 16 + vg * 2;   // A frag k map
      a0[2*vg] = ar0[ka]; a0[2*vg+1] = ar0[ka+1];
      a1[2*vg] = ar1[ka]; a1[2*vg+1] = ar1[ka+1];
      int kb = hid * 16 + vg * 2;                                  // B frag k map
      b0[2*vg] = br0[kb]; b0[2*vg+1] = br0[kb+1];
      b1[2*vg] = br1[kb]; b1[2*vg+1] = br1[kb+1];
    }
    acc00 = __builtin_amdgcn_wmma_f32_16x16x32_f16(false, a0, false, b0, (short)0, acc00, false, false);
    acc01 = __builtin_amdgcn_wmma_f32_16x16x32_f16(false, a0, false, b1, (short)0, acc01, false, false);
    acc10 = __builtin_amdgcn_wmma_f32_16x16x32_f16(false, a1, false, b0, (short)0, acc10, false, false);
    acc11 = __builtin_amdgcn_wmma_f32_16x16x32_f16(false, a1, false, b1, (short)0, acc11, false, false);

#if HAS_ASYNC_LDS
    wait_async0();
#endif
    __syncthreads();
  }

  const int nA = n0 + tn + lrow;
  const int nB = nA + 16;
  const float bvA = bias[nA];
  const float bvB = bias[nB];
#pragma unroll
  for (int r = 0; r < 8; ++r) {
    int mA = m0 + tm + r + hid * 8;
    int mB = mA + 16;
    float v00 = acc00[r] + bvA;
    float v01 = acc01[r] + bvB;
    float v10 = acc10[r] + bvA;
    float v11 = acc11[r] + bvB;
    if (act == 1) {
      v00 = fmaxf(v00, 0.0f); v01 = fmaxf(v01, 0.0f);
      v10 = fmaxf(v10, 0.0f); v11 = fmaxf(v11, 0.0f);
    }
    if (C32) {
      C32[(size_t)mA * N + nA] = v00; C32[(size_t)mA * N + nB] = v01;
      C32[(size_t)mB * N + nA] = v10; C32[(size_t)mB * N + nB] = v11;
    }
    if (C16) {
      C16[(size_t)mA * N + nA] = (_Float16)v00; C16[(size_t)mA * N + nB] = (_Float16)v01;
      C16[(size_t)mB * N + nA] = (_Float16)v10; C16[(size_t)mB * N + nB] = (_Float16)v11;
    }
  }
}

// ---------------------------------------------------------------------------
// Fused flash attention: one wave per (16-query tile, head).
// ---------------------------------------------------------------------------
__global__ __launch_bounds__(32)
void flash_attn_kernel(const _Float16* __restrict__ Q, const _Float16* __restrict__ Kt,
                       const _Float16* __restrict__ V, _Float16* __restrict__ O) {
  __shared__ __align__(16) _Float16 pl[16 * 48];
  const int lane = threadIdx.x;
  const int lrow = lane & 15;
  const int hid  = lane >> 4;
  const int q0 = blockIdx.x * 16;
  const int hc = blockIdx.y * HDIM;      // head column base
  const float scale = 0.125f;            // 1/sqrt(64)

  v16h aq0, aq1;
#pragma unroll
  for (int vg = 0; vg < 8; ++vg) {
    int kb = hid * 16 + vg * 2;
    const _Float16* p = Q + (size_t)(q0 + lrow) * EMB + hc + kb;
    aq0[2 * vg] = p[0];  aq0[2 * vg + 1] = p[1];
    aq1[2 * vg] = p[32]; aq1[2 * vg + 1] = p[33];
  }
  float rmax[8], rsum[8];
  v8f oac[4];
#pragma unroll
  for (int r = 0; r < 8; ++r) { rmax[r] = -1e30f; rsum[r] = 0.0f; }
#pragma unroll
  for (int g = 0; g < 4; ++g) oac[g] = (v8f){};

  for (int kb = 0; kb < SEQ; kb += 32) {
    v16h b;
    v8f s0 = {}, s1 = {};
    const _Float16* kp0 = Kt + (size_t)(kb + lrow) * EMB + hc;
#pragma unroll
    for (int vg = 0; vg < 8; ++vg) { int kd = hid * 16 + vg * 2; b[2*vg] = kp0[kd]; b[2*vg+1] = kp0[kd+1]; }
    s0 = __builtin_amdgcn_wmma_f32_16x16x32_f16(false, aq0, false, b, (short)0, s0, false, false);
#pragma unroll
    for (int vg = 0; vg < 8; ++vg) { int kd = 32 + hid * 16 + vg * 2; b[2*vg] = kp0[kd]; b[2*vg+1] = kp0[kd+1]; }
    s0 = __builtin_amdgcn_wmma_f32_16x16x32_f16(false, aq1, false, b, (short)0, s0, false, false);
    const _Float16* kp1 = Kt + (size_t)(kb + 16 + lrow) * EMB + hc;
#pragma unroll
    for (int vg = 0; vg < 8; ++vg) { int kd = hid * 16 + vg * 2; b[2*vg] = kp1[kd]; b[2*vg+1] = kp1[kd+1]; }
    s1 = __builtin_amdgcn_wmma_f32_16x16x32_f16(false, aq0, false, b, (short)0, s1, false, false);
#pragma unroll
    for (int vg = 0; vg < 8; ++vg) { int kd = 32 + hid * 16 + vg * 2; b[2*vg] = kp1[kd]; b[2*vg+1] = kp1[kd+1]; }
    s1 = __builtin_amdgcn_wmma_f32_16x16x32_f16(false, aq1, false, b, (short)0, s1, false, false);

    float tmx[8], ps[8];
#pragma unroll
    for (int r = 0; r < 8; ++r) {
      float a0 = s0[r] * scale, a1 = s1[r] * scale;
      s0[r] = a0; s1[r] = a1;
      tmx[r] = fmaxf(a0, a1);
    }
#pragma unroll
    for (int off = 8; off >= 1; off >>= 1) {
#pragma unroll
      for (int r = 0; r < 8; ++r) tmx[r] = fmaxf(tmx[r], __shfl_xor(tmx[r], off, 32));
    }
#pragma unroll
    for (int r = 0; r < 8; ++r) {
      float nm = fmaxf(rmax[r], tmx[r]);
      float al = __expf(rmax[r] - nm);
      rmax[r] = nm;
      rsum[r] *= al;
      oac[0][r] *= al; oac[1][r] *= al; oac[2][r] *= al; oac[3][r] *= al;
      float p0 = __expf(s0[r] - nm);
      float p1 = __expf(s1[r] - nm);
      ps[r] = p0 + p1;
      int m = r + hid * 8;
      pl[m * 48 + lrow]      = (_Float16)p0;
      pl[m * 48 + 16 + lrow] = (_Float16)p1;
    }
#pragma unroll
    for (int off = 8; off >= 1; off >>= 1) {
#pragma unroll
      for (int r = 0; r < 8; ++r) ps[r] += __shfl_xor(ps[r], off, 32);
    }
#pragma unroll
    for (int r = 0; r < 8; ++r) rsum[r] += ps[r];

    asm volatile("s_wait_dscnt 0" ::: "memory");   // P tile visible to whole wave

    v16h pa;
    const _Float16* prow = pl + lrow * 48;
#pragma unroll
    for (int vg = 0; vg < 8; ++vg) {
      int kbp = hid * 16 + vg * 2;
      pa[2 * vg] = prow[kbp]; pa[2 * vg + 1] = prow[kbp + 1];
    }
#pragma unroll
    for (int g = 0; g < 4; ++g) {
      v16h bv;
#pragma unroll
      for (int vg = 0; vg < 8; ++vg) {
        int ko = hid * 16 + vg * 2;
        bv[2 * vg]     = V[(size_t)(kb + ko) * EMB + hc + g * 16 + lrow];
        bv[2 * vg + 1] = V[(size_t)(kb + ko + 1) * EMB + hc + g * 16 + lrow];
      }
      oac[g] = __builtin_amdgcn_wmma_f32_16x16x32_f16(false, pa, false, bv, (short)0, oac[g], false, false);
    }
    asm volatile("s_wait_dscnt 0" ::: "memory");   // reads done before next tile's stores
  }
#pragma unroll
  for (int r = 0; r < 8; ++r) {
    float inv = 1.0f / rsum[r];
    int m = q0 + r + hid * 8;
#pragma unroll
    for (int g = 0; g < 4; ++g)
      O[(size_t)m * EMB + hc + g * 16 + lrow] = (_Float16)(oac[g][r] * inv);
  }
}

// ---------------------------------------------------------------------------
// residual add + layernorm over D=512 (one block per token)
// ---------------------------------------------------------------------------
__global__ __launch_bounds__(256)
void add_ln_kernel(const float* __restrict__ x, const float* __restrict__ d,
                   const float* __restrict__ g, const float* __restrict__ b,
                   float* __restrict__ outf, _Float16* __restrict__ outh) {
  __shared__ float red[256];
  int row = blockIdx.x, t = threadIdx.x;
  const float* xr = x + (size_t)row * EMB;
  const float* dr = d + (size_t)row * EMB;
  float v0 = xr[t] + dr[t];
  float v1 = xr[t + 256] + dr[t + 256];
  red[t] = v0 + v1;
  __syncthreads();
  for (int off = 128; off > 0; off >>= 1) { if (t < off) red[t] += red[t + off]; __syncthreads(); }
  float mean = red[0] * (1.0f / 512.0f);
  __syncthreads();
  float e0 = v0 - mean, e1 = v1 - mean;
  red[t] = e0 * e0 + e1 * e1;
  __syncthreads();
  for (int off = 128; off > 0; off >>= 1) { if (t < off) red[t] += red[t + off]; __syncthreads(); }
  float inv = rsqrtf(red[0] * (1.0f / 512.0f) + 1e-5f);
  float o0 = e0 * inv * g[t] + b[t];
  float o1 = e1 * inv * g[t + 256] + b[t + 256];
  size_t base = (size_t)row * EMB;
  outf[base + t] = o0;           outf[base + t + 256] = o1;
  outh[base + t] = (_Float16)o0; outh[base + t + 256] = (_Float16)o1;
}

// ---------------------------------------------------------------------------
// heads: disp*constraint, quality MLP, warp
// ---------------------------------------------------------------------------
__global__ void dispmul_kernel(const float* __restrict__ draw, const float* __restrict__ cons,
                               float* __restrict__ dout) {
  int i = blockIdx.x * blockDim.x + threadIdx.x;
  if (i >= 2 * NPIX) return;
  dout[i] = draw[i] * cons[i % NPIX];
}

__global__ __launch_bounds__(256)
void qual1_kernel(const float* __restrict__ df, const float* __restrict__ w,
                  const float* __restrict__ b, const float* __restrict__ g,
                  const float* __restrict__ be, float* __restrict__ out) {
  __shared__ float red[256];
  int j = threadIdx.x;
  float acc = b[j];
  for (int i = 0; i < 2 * NPIX; ++i) acc = __builtin_fmaf(df[i], w[(size_t)i * 256 + j], acc);
  red[j] = acc;
  __syncthreads();
  for (int off = 128; off > 0; off >>= 1) { if (j < off) red[j] += red[j + off]; __syncthreads(); }
  float mean = red[0] * (1.0f / 256.0f);
  __syncthreads();
  float e = acc - mean;
  red[j] = e * e;
  __syncthreads();
  for (int off = 128; off > 0; off >>= 1) { if (j < off) red[j] += red[j + off]; __syncthreads(); }
  float inv = rsqrtf(red[0] * (1.0f / 256.0f) + 1e-5f);
  out[j] = fmaxf(e * inv * g[j] + be[j], 0.0f);
}

__global__ void fc_relu_kernel(const float* __restrict__ in, const float* __restrict__ w,
                               const float* __restrict__ b, float* __restrict__ out,
                               int In, int Out) {
  int j = blockIdx.x * blockDim.x + threadIdx.x;
  if (j >= Out) return;
  float acc = b[j];
  for (int i = 0; i < In; ++i) acc = __builtin_fmaf(in[i], w[i * Out + j], acc);
  out[j] = fmaxf(acc, 0.0f);
}

__global__ __launch_bounds__(64)
void qual_final_kernel(const float* __restrict__ in, const float* __restrict__ w,
                       const float* __restrict__ b, float* __restrict__ out) {
  __shared__ float red[64];
  int t = threadIdx.x;
  red[t] = in[t] * w[t];
  __syncthreads();
  for (int off = 32; off > 0; off >>= 1) { if (t < off) red[t] += red[t + off]; __syncthreads(); }
  if (t == 0) out[0] = 1.0f / (1.0f + __expf(-(red[0] + b[0])));
}

__global__ void warp_kernel(const float* __restrict__ img1, const float* __restrict__ img2,
                            const float* __restrict__ disp, float* __restrict__ matching) {
  int idx = blockIdx.x * blockDim.x + threadIdx.x;
  if (idx >= 3 * NPIX) return;
  int c = idx / NPIX, p = idx - c * NPIX;
  int yh = p / W_IMG, xw = p - yh * W_IMG;
  float lx = -1.0f + 2.0f * (float)xw / 47.0f;
  float ly = -1.0f + 2.0f * (float)yh / 47.0f;
  float wgx = fminf(fmaxf(lx + disp[p],        -1.0f), 1.0f);
  float wgy = fminf(fmaxf(ly + disp[NPIX + p], -1.0f), 1.0f);
  float ix = fminf(fmaxf(((wgx + 1.0f) * 48.0f - 1.0f) * 0.5f, 0.0f), 47.0f);
  float iy = fminf(fmaxf(((wgy + 1.0f) * 48.0f - 1.0f) * 0.5f, 0.0f), 47.0f);
  int x0 = (int)floorf(ix), y0 = (int)floorf(iy);
  float wx = ix - (float)x0, wy = iy - (float)y0;
  int x1 = min(x0 + 1, 47), y1 = min(y0 + 1, 47);
  const float* im = img2 + (size_t)c * NPIX;
  float v00 = im[y0 * W_IMG + x0], v01 = im[y0 * W_IMG + x1];
  float v10 = im[y1 * W_IMG + x0], v11 = im[y1 * W_IMG + x1];
  float wv = v00 * (1.0f - wx) * (1.0f - wy) + v01 * wx * (1.0f - wy)
           + v10 * (1.0f - wx) * wy + v11 * wx * wy;
  matching[idx] = (img1[idx] + wv) * 0.5f;
}

// ---------------------------------------------------------------------------
// host launcher
// ---------------------------------------------------------------------------
static inline int cdiv(int a, int b) { return (a + b - 1) / b; }

extern "C" void kernel_launch(void* const* d_in, const int* in_sizes, int n_in,
                              void* d_out, int out_size, void* d_ws, size_t ws_size,
                              hipStream_t stream) {
  (void)in_sizes; (void)n_in; (void)out_size; (void)ws_size;
  const float* image1 = (const float*)d_in[0];
  const float* image2 = (const float*)d_in[1];
  // params flattened by sorted dict keys: cons, emb, layers, outp, qual
  const float* cons_b[3] = {(const float*)d_in[2], (const float*)d_in[4], (const float*)d_in[6]};
  const float* cons_w[3] = {(const float*)d_in[3], (const float*)d_in[5], (const float*)d_in[7]};
  const float* emb_b[3]  = {(const float*)d_in[8], (const float*)d_in[10], (const float*)d_in[12]};
  const float* emb_w[3]  = {(const float*)d_in[9], (const float*)d_in[11], (const float*)d_in[13]};
  const int LBASE = 14;
  const int OBASE = 14 + 6 * 16;   // 110
  const float* outp_b[3] = {(const float*)d_in[OBASE+0], (const float*)d_in[OBASE+2], (const float*)d_in[OBASE+4]};
  const float* outp_w[3] = {(const float*)d_in[OBASE+1], (const float*)d_in[OBASE+3], (const float*)d_in[OBASE+5]};
  const int QBASE = OBASE + 6;     // 116: b1,b2,b3,b4,be,g,w1,w2,w3,w4
  const float* q_b1 = (const float*)d_in[QBASE+0];
  const float* q_b2 = (const float*)d_in[QBASE+1];
  const float* q_b3 = (const float*)d_in[QBASE+2];
  const float* q_b4 = (const float*)d_in[QBASE+3];
  const float* q_be = (const float*)d_in[QBASE+4];
  const float* q_g  = (const float*)d_in[QBASE+5];
  const float* q_w1 = (const float*)d_in[QBASE+6];
  const float* q_w2 = (const float*)d_in[QBASE+7];
  const float* q_w3 = (const float*)d_in[QBASE+8];
  const float* q_w4 = (const float*)d_in[QBASE+9];

  // output regions
  float* out = (float*)d_out;
  float* out_disp = out;                    // 4608
  float* out_match = out + 4608;            // 6912
  float* out_qual = out + 11520;            // 1
  float* out_cons = out + 11521;            // 2304
  float* out_feat = out + 13825;            // 512*2304

  // workspace carve-up
  char* wp = (char*)d_ws;
  auto carve = [&](size_t bytes) -> void* {
    void* r = (void*)wp;
    wp += (bytes + 255) & ~(size_t)255;
    return r;
  };
  float*    t_f   = (float*)   carve((size_t)SEQ * EMB * 4);
  _Float16* t_h   = (_Float16*)carve((size_t)SEQ * EMB * 2);
  _Float16* qh_   = (_Float16*)carve((size_t)SEQ * EMB * 2);
  _Float16* kh_   = (_Float16*)carve((size_t)SEQ * EMB * 2);
  _Float16* vh_   = (_Float16*)carve((size_t)SEQ * EMB * 2);
  _Float16* ao_h  = (_Float16*)carve((size_t)SEQ * EMB * 2);
  float*    tmp_f = (float*)   carve((size_t)SEQ * EMB * 4);
  _Float16* hid_h = (_Float16*)carve((size_t)SEQ * FF_DIM * 2);
  _Float16* wbuf  = (_Float16*)carve((size_t)EMB * FF_DIM * 2);
  float*    x0    = (float*)   carve((size_t)6  * NPIX * 4);
  float*    b64a  = (float*)   carve((size_t)64 * NPIX * 4);
  float*    b128  = (float*)   carve((size_t)128* NPIX * 4);
  float*    b32   = (float*)   carve((size_t)32 * NPIX * 4);
  float*    draw  = (float*)   carve((size_t)2  * NPIX * 4);
  float*    qtmp  = (float*)   carve(512 * 4);   // 256 + 128 + 64
  float* qt1 = qtmp, *qt2 = qtmp + 256, *qt3 = qtmp + 384;

  // ---- conv stem ----
  concat_kernel<<<cdiv(3 * NPIX, 256), 256, 0, stream>>>(image1, image2, x0);
  conv3x3_kernel<<<cdiv(64  * NPIX, 256), 256, 0, stream>>>(x0,   emb_w[0], emb_b[0], b64a, 6,   64,  1);
  conv3x3_kernel<<<cdiv(128 * NPIX, 256), 256, 0, stream>>>(b64a, emb_w[1], emb_b[1], b128, 64,  128, 1);
  conv3x3_kernel<<<cdiv(512 * NPIX, 256), 256, 0, stream>>>(b128, emb_w[2], emb_b[2], tmp_f,128, 512, 1);
  posenc_transpose_kernel<<<cdiv(SEQ * EMB, 256), 256, 0, stream>>>(tmp_f, t_f, t_h);

  // ---- transformer layers ----
  const dim3 gemmD(EMB / GBN, SEQ / GBM);      // (4, 36)
  const dim3 gemmF(FF_DIM / GBN, SEQ / GBM);   // (16, 36)
  for (int l = 0; l < 6; ++l) {
    const float* b1  = (const float*)d_in[LBASE + l * 16 + 0];
    const float* b2  = (const float*)d_in[LBASE + l * 16 + 1];
    const float* be1 = (const float*)d_in[LBASE + l * 16 + 2];
    const float* be2 = (const float*)d_in[LBASE + l * 16 + 3];
    const float* bk  = (const float*)d_in[LBASE + l * 16 + 4];
    const float* bo  = (const float*)d_in[LBASE + l * 16 + 5];
    const float* bq  = (const float*)d_in[LBASE + l * 16 + 6];
    const float* bvv = (const float*)d_in[LBASE + l * 16 + 7];
    const float* g1  = (const float*)d_in[LBASE + l * 16 + 8];
    const float* g2  = (const float*)d_in[LBASE + l * 16 + 9];
    const float* w1  = (const float*)d_in[LBASE + l * 16 + 10];
    const float* w2  = (const float*)d_in[LBASE + l * 16 + 11];
    const float* wk  = (const float*)d_in[LBASE + l * 16 + 12];
    const float* wo  = (const float*)d_in[LBASE + l * 16 + 13];
    const float* wq  = (const float*)d_in[LBASE + l * 16 + 14];
    const float* wv  = (const float*)d_in[LBASE + l * 16 + 15];

    f2h_t_kernel<<<cdiv(EMB * EMB, 256), 256, 0, stream>>>(wq, wbuf, EMB, EMB);
    gemm_wmma_kernel<<<gemmD, 256, 0, stream>>>(t_h, wbuf, bq, nullptr, qh_, SEQ, EMB, EMB, 0);
    f2h_t_kernel<<<cdiv(EMB * EMB, 256), 256, 0, stream>>>(wk, wbuf, EMB, EMB);
    gemm_wmma_kernel<<<gemmD, 256, 0, stream>>>(t_h, wbuf, bk, nullptr, kh_, SEQ, EMB, EMB, 0);
    f2h_t_kernel<<<cdiv(EMB * EMB, 256), 256, 0, stream>>>(wv, wbuf, EMB, EMB);
    gemm_wmma_kernel<<<gemmD, 256, 0, stream>>>(t_h, wbuf, bvv, nullptr, vh_, SEQ, EMB, EMB, 0);

    flash_attn_kernel<<<dim3(SEQ / 16, NHEAD), 32, 0, stream>>>(qh_, kh_, vh_, ao_h);

    f2h_t_kernel<<<cdiv(EMB * EMB, 256), 256, 0, stream>>>(wo, wbuf, EMB, EMB);
    gemm_wmma_kernel<<<gemmD, 256, 0, stream>>>(ao_h, wbuf, bo, tmp_f, nullptr, SEQ, EMB, EMB, 0);
    add_ln_kernel<<<SEQ, 256, 0, stream>>>(t_f, tmp_f, g1, be1, t_f, t_h);

    f2h_t_kernel<<<cdiv(EMB * FF_DIM, 256), 256, 0, stream>>>(w1, wbuf, EMB, FF_DIM);
    gemm_wmma_kernel<<<gemmF, 256, 0, stream>>>(t_h, wbuf, b1, nullptr, hid_h, SEQ, FF_DIM, EMB, 1);
    f2h_t_kernel<<<cdiv(EMB * FF_DIM, 256), 256, 0, stream>>>(w2, wbuf, FF_DIM, EMB);
    gemm_wmma_kernel<<<gemmD, 256, 0, stream>>>(hid_h, wbuf, b2, tmp_f, nullptr, SEQ, EMB, FF_DIM, 0);
    add_ln_kernel<<<SEQ, 256, 0, stream>>>(t_f, tmp_f, g2, be2, t_f, t_h);
  }

  // ---- features output (D,H,W) ----
  feat_transpose_kernel<<<cdiv(SEQ * EMB, 256), 256, 0, stream>>>(t_f, out_feat);

  // ---- output head convs ----
  conv3x3_kernel<<<cdiv(128 * NPIX, 256), 256, 0, stream>>>(out_feat, outp_w[0], outp_b[0], b128, 512, 128, 1);
  conv3x3_kernel<<<cdiv(64  * NPIX, 256), 256, 0, stream>>>(b128,     outp_w[1], outp_b[1], b64a, 128, 64,  1);
  conv3x3_kernel<<<cdiv(2   * NPIX, 256), 256, 0, stream>>>(b64a,     outp_w[2], outp_b[2], draw, 64,  2,   2);

  // ---- constraint convs ----
  conv3x3_kernel<<<cdiv(64 * NPIX, 256), 256, 0, stream>>>(draw, cons_w[0], cons_b[0], b128, 2,  64, 1);
  conv3x3_kernel<<<cdiv(32 * NPIX, 256), 256, 0, stream>>>(b128, cons_w[1], cons_b[1], b32,  64, 32, 1);
  conv3x3_kernel<<<cdiv(1  * NPIX, 256), 256, 0, stream>>>(b32,  cons_w[2], cons_b[2], out_cons, 32, 1, 3);

  dispmul_kernel<<<cdiv(2 * NPIX, 256), 256, 0, stream>>>(draw, out_cons, out_disp);

  // ---- quality MLP (on final disp) ----
  qual1_kernel<<<1, 256, 0, stream>>>(out_disp, q_w1, q_b1, q_g, q_be, qt1);
  fc_relu_kernel<<<1, 128, 0, stream>>>(qt1, q_w2, q_b2, qt2, 256, 128);
  fc_relu_kernel<<<1, 64,  0, stream>>>(qt2, q_w3, q_b3, qt3, 128, 64);
  qual_final_kernel<<<1, 64, 0, stream>>>(qt3, q_w4, q_b4, out_qual);

  // ---- warp + matching ----
  warp_kernel<<<cdiv(3 * NPIX, 256), 256, 0, stream>>>(image1, image2, out_disp, out_match);
}
